// LorentzNet_39195871543560
// MI455X (gfx1250) — matrix operand order repaired and checked by reading, not compiled, not run on hardware
//
#include <hip/hip_runtime.h>

// ---------------------------------------------------------------------------
// LorentzNet on gfx1250: bf16 WMMA (16x16x32) edge GEMMs, f32 accumulation.
// Weight panels LDS-resident; B fragments double-buffered (one-deep software
// pipeline) so each v_wmma waits only on loads issued a full iteration ahead
// (s_wait_dscnt <= 2 instead of 0).
// ---------------------------------------------------------------------------

typedef __attribute__((ext_vector_type(16))) __bf16 v16bf;
typedef __attribute__((ext_vector_type(8)))  float  v8f;

#define N_NODES 20000
#define N_EDGES 320000
#define HID     128
#define D_IN    262
#define NTILES  8     // 128 / 16 output tiles
#define KCHUNKS 4     // 128 / 32 K chunks
#define BN_EPS  1e-5f
#define WMAT    16384 // elements per swizzled 128x128 weight panel

union Frag16 {
    v16bf v;
    uint4 u[2];
    unsigned short s[16];
};

static __device__ __forceinline__ unsigned short f2bf(float f) {
    unsigned int x = __float_as_uint(f);
    unsigned int r = x + 0x7fffu + ((x >> 16) & 1u);  // round-to-nearest-even
    return (unsigned short)(r >> 16);
}

static __device__ __forceinline__ float psi_f(float z) {
    return copysignf(log1pf(fabsf(z)), z);
}

static __device__ __forceinline__ float sigmoid_f(float z) {
    return 1.f / (1.f + __expf(-z));
}

// load the swizzled B fragment (ntile n, chunk c) for this lane from an
// LDS-resident panel
static __device__ __forceinline__ void ldB(Frag16& d, const unsigned short* base,
                                           int c, int n, int lane) {
    const unsigned short* bp = base + ((n * KCHUNKS + c) * 32 + lane) * 16;
    d.u[0] = *(const uint4*)(bp);
    d.u[1] = *(const uint4*)(bp + 8);
}

// ---------------------------------------------------------------------------
// Weight prep: convert W1a/W1b/W2/Wa (per layer) to bf16 in WMMA-B fragment
// order so GEMM kernels load contiguous 32B per lane.
// ---------------------------------------------------------------------------
__global__ void k_prep_weights(const float* __restrict__ W1,
                               const float* __restrict__ W2,
                               const float* __restrict__ Wa,
                               unsigned short* __restrict__ wsw) {
    int t = blockIdx.x * blockDim.x + threadIdx.x;
    if (t >= 3 * 4 * WMAT) return;
    int within = t & (WMAT - 1);
    int mat    = t >> 14;          // 0..11  (layer*4 + m)
    int l      = mat >> 2;
    int m      = mat & 3;
    int i      = within & 15;
    int lane   = (within >> 4) & 31;
    int chunk  = (within >> 9) & 3;
    int ntile  = within >> 11;
    int kRow   = chunk * 32 + (i >> 3) * 16 + (lane >> 4) * 8 + (i & 7);
    int nCol   = ntile * 16 + (lane & 15);
    const float* src;
    if      (m == 0) src = W1 + (size_t)l * D_IN * HID;              // rows 0..127   (x_i)
    else if (m == 1) src = W1 + (size_t)l * D_IN * HID + 128 * HID;  // rows 128..255 (x_j)
    else if (m == 2) src = W2 + (size_t)l * HID * HID;
    else             src = Wa + (size_t)l * HID * HID;
    wsw[t] = f2bf(src[kRow * HID + nCol]);
}

// ---------------------------------------------------------------------------
// Per-layer prologue: x_next = x_cur, xbf = bf16(x_cur), zero BN stats.
// ---------------------------------------------------------------------------
__global__ void k_copy(const float* __restrict__ cur, float* __restrict__ nxt,
                       unsigned short* __restrict__ xbf, float* __restrict__ stats) {
    int gid = blockIdx.x * blockDim.x + threadIdx.x;
    if (gid < 512) stats[gid] = 0.f;
    const int total = N_NODES * HID;
    for (int i = gid; i < total; i += gridDim.x * blockDim.x) {
        float v = cur[i];
        nxt[i]  = v;
        xbf[i]  = f2bf(v);
    }
}

// ---------------------------------------------------------------------------
// Pass 1: hpre = concat(x_i, x_j, edge_attr, norms, dots) @ W1  (per edge),
// plus per-channel sum / sumsq for BatchNorm.
// One wave per 16-edge tile; W1a/W1b panels LDS-resident; B pipelined.
// ---------------------------------------------------------------------------
__global__ void __launch_bounds__(128) k_edge_pass1(
    const unsigned short* __restrict__ xbf, const float* __restrict__ xc,
    const int* __restrict__ row, const int* __restrict__ col,
    const float* __restrict__ eattr,
    const unsigned short* __restrict__ W1a_sw, const unsigned short* __restrict__ W1b_sw,
    const float* __restrict__ W1c,                      // [6][128] f32 tail
    float* __restrict__ hpre,
    float* __restrict__ stat_sum, float* __restrict__ stat_sq)
{
    __shared__ __attribute__((aligned(16))) unsigned short wA[WMAT];
    __shared__ __attribute__((aligned(16))) unsigned short wB[WMAT];
    __shared__ float w1c_l[6 * HID];
    __shared__ float ls_sum[HID];
    __shared__ float ls_sq[HID];
    __shared__ float sm[4][16][6];

    int tid = threadIdx.x;
    // stage weight panels + tail + zero stats
    {
        const uint4* sA = (const uint4*)W1a_sw;
        const uint4* sB = (const uint4*)W1b_sw;
        uint4* dA = (uint4*)wA;
        uint4* dB = (uint4*)wB;
        #pragma unroll
        for (int i = 0; i < WMAT / 8 / 128; ++i) {      // 2048 uint4 / 128 thr = 16
            dA[tid + i * 128] = sA[tid + i * 128];
            dB[tid + i * 128] = sB[tid + i * 128];
        }
        #pragma unroll
        for (int i = tid; i < 6 * HID; i += 128) w1c_l[i] = W1c[i];
        if (tid < HID) { ls_sum[tid] = 0.f; ls_sq[tid] = 0.f; }
    }
    __syncthreads();

    int wv = tid >> 5, lane = tid & 31, half = lane >> 4, ml = lane & 15;
    const int NT = N_EDGES / 16;

    for (int t = blockIdx.x * 4 + wv; t < NT; t += gridDim.x * 4) {
        int e0 = t * 16;
        int e  = e0 + ml;
        int nI = row[e];
        int nJ = col[e];

        // ---- scalar phase: Minkowski dots + edge scalars into LDS ----
        {
            const float* xi = xc + (size_t)nI * HID + half * 64;
            const float* xj = xc + (size_t)nJ * HID + half * 64;
            float pii = 0.f, pjj = 0.f, pij = 0.f;
            #pragma unroll
            for (int k4 = 0; k4 < 16; ++k4) {
                float4 a = ((const float4*)xi)[k4];
                float4 b = ((const float4*)xj)[k4];
                pii += a.x * a.x + a.y * a.y + a.z * a.z + a.w * a.w;
                pjj += b.x * b.x + b.y * b.y + b.z * b.z + b.w * b.w;
                pij += a.x * b.x + a.y * b.y + a.z * b.z + a.w * b.w;
            }
            // mink = a0*b0 - sum_{k>=1} = -(plain) + 2*a0*b0 (k=0 lives in half 0)
            float mii = -pii, mjj = -pjj, mij = -pij;
            if (half == 0) {
                float a0 = xi[0], b0 = xj[0];
                mii += 2.f * a0 * a0;
                mjj += 2.f * b0 * b0;
                mij += 2.f * a0 * b0;
            }
            mii += __shfl_xor(mii, 16, 32);
            mjj += __shfl_xor(mjj, 16, 32);
            mij += __shfl_xor(mij, 16, 32);
            float dd = mii - 2.f * mij + mjj;          // mink(x_diff, x_diff)
            if (half == 0) {
                float4 ea = *(const float4*)(eattr + (size_t)e * 4);
                sm[wv][ml][0] = ea.x; sm[wv][ml][1] = ea.y;
                sm[wv][ml][2] = ea.z; sm[wv][ml][3] = ea.w;
                sm[wv][ml][4] = psi_f(dd);
                sm[wv][ml][5] = psi_f(mij);
            }
        }
        asm volatile("s_wait_dscnt 0" ::: "memory");

        // ---- WMMA: acc = x_i @ W1a + x_j @ W1b, B pipelined one deep ----
        v8f acc[NTILES] = {};
        const unsigned short* xbI = xbf + (size_t)nI * HID;
        const unsigned short* xbJ = xbf + (size_t)nJ * HID;
        Frag16 bb[2];
        ldB(bb[0], wA, 0, 0, lane);
        #pragma unroll
        for (int cc = 0; cc < 2 * KCHUNKS; ++cc) {      // 0..3: x_i/W1a, 4..7: x_j/W1b
            const unsigned short* xb = (cc < KCHUNKS) ? xbI : xbJ;
            int c = cc & 3;
            Frag16 a;
            a.u[0] = *(const uint4*)(xb + c * 32 + half * 8);
            a.u[1] = *(const uint4*)(xb + c * 32 + 16 + half * 8);
            #pragma unroll
            for (int n = 0; n < NTILES; ++n) {
                int idx = cc * NTILES + n;
                if (idx + 1 < 2 * KCHUNKS * NTILES) {
                    int i2 = idx + 1;
                    const unsigned short* wbase = (i2 < KCHUNKS * NTILES) ? wA : wB;
                    ldB(bb[i2 & 1], wbase, (i2 >> 3) & 3, i2 & 7, lane);
                }
                acc[n] = __builtin_amdgcn_wmma_f32_16x16x32_bf16(
                    false, a.v, false, bb[idx & 1].v, (short)0, acc[n], false, false);
            }
        }

        // ---- rank-6 tail (edge_attr, norms, dots) in f32 ----
        #pragma unroll
        for (int r = 0; r < 8; ++r) {
            int M = half * 8 + r;
            float s0 = sm[wv][M][0], s1 = sm[wv][M][1], s2 = sm[wv][M][2];
            float s3 = sm[wv][M][3], s4 = sm[wv][M][4], s5 = sm[wv][M][5];
            #pragma unroll
            for (int n = 0; n < NTILES; ++n) {
                int N = n * 16 + ml;
                acc[n][r] += s0 * w1c_l[0 * HID + N] + s1 * w1c_l[1 * HID + N]
                           + s2 * w1c_l[2 * HID + N] + s3 * w1c_l[3 * HID + N]
                           + s4 * w1c_l[4 * HID + N] + s5 * w1c_l[5 * HID + N];
            }
        }

        // ---- store hpre + accumulate BN stats ----
        #pragma unroll
        for (int n = 0; n < NTILES; ++n) {
            int N = n * 16 + ml;
            float cs = 0.f, cq = 0.f;
            #pragma unroll
            for (int r = 0; r < 8; ++r) {
                float v = acc[n][r];
                hpre[(size_t)(e0 + half * 8 + r) * HID + N] = v;
                cs += v;
                cq += v * v;
            }
            cs += __shfl_xor(cs, 16, 32);
            cq += __shfl_xor(cq, 16, 32);
            if (half == 0) {
                atomicAdd(&ls_sum[N], cs);
                atomicAdd(&ls_sq[N], cq);
            }
        }
    }

    __syncthreads();
    if (tid < HID) {
        atomicAdd(&stat_sum[tid], ls_sum[tid]);
        atomicAdd(&stat_sq[tid],  ls_sq[tid]);
    }
}

// ---------------------------------------------------------------------------
// Fold BN stats + gamma/beta into per-channel scale/bias.
// ---------------------------------------------------------------------------
__global__ void k_bn(const float* __restrict__ stat_sum, const float* __restrict__ stat_sq,
                     const float* __restrict__ gamma, const float* __restrict__ beta,
                     float* __restrict__ scale, float* __restrict__ bias) {
    int c = threadIdx.x;
    float inv = 1.f / (float)N_EDGES;
    float mu  = stat_sum[c] * inv;
    float var = stat_sq[c] * inv - mu * mu;
    float sc  = gamma[c] * rsqrtf(var + BN_EPS);
    scale[c]  = sc;
    bias[c]   = beta[c] - mu * sc;
}

// ---------------------------------------------------------------------------
// Pass 2: BN+ReLU -> WMMA(W2)+b2 -> ReLU -> sigmoid gate (Wm,bm) ->
// LDS transpose -> WMMA(Wa)+ba -> ReLU -> dot(Wb) = s ->
// scatter atomicAdd(clip(x_diff * s)).  W2/Wa panels LDS-resident; B pipelined.
// ---------------------------------------------------------------------------
__global__ void __launch_bounds__(128) k_edge_pass2(
    const float* __restrict__ hpre,
    const float* __restrict__ scale, const float* __restrict__ bias,
    const unsigned short* __restrict__ W2_sw, const unsigned short* __restrict__ Wa_sw,
    const float* __restrict__ b2, const float* __restrict__ ba,
    const float* __restrict__ Wm, const float* __restrict__ bm,
    const float* __restrict__ Wb,
    const float* __restrict__ xc,
    const int* __restrict__ row, const int* __restrict__ col,
    float* __restrict__ xnxt)
{
    __shared__ __attribute__((aligned(16))) unsigned short w2l[WMAT];
    __shared__ __attribute__((aligned(16))) unsigned short wal[WMAT];
    __shared__ __attribute__((aligned(16))) unsigned short mt[4][16 * HID];
    __shared__ float sc_l[HID];
    __shared__ float bi_l[HID];
    __shared__ float sS[4][16];

    int tid = threadIdx.x, wv = tid >> 5, lane = tid & 31, half = lane >> 4, ml = lane & 15;

    // stage weight panels + BN scale/bias
    {
        const uint4* s2 = (const uint4*)W2_sw;
        const uint4* sa = (const uint4*)Wa_sw;
        uint4* d2 = (uint4*)w2l;
        uint4* da = (uint4*)wal;
        #pragma unroll
        for (int i = 0; i < WMAT / 8 / 128; ++i) {
            d2[tid + i * 128] = s2[tid + i * 128];
            da[tid + i * 128] = sa[tid + i * 128];
        }
        if (tid < HID) { sc_l[tid] = scale[tid]; bi_l[tid] = bias[tid]; }
    }
    __syncthreads();

    // tile-invariant per-lane params
    float bmv = bm[0];
    float b2v[NTILES], bav[NTILES], wmv[NTILES], wbv[NTILES];
    #pragma unroll
    for (int n = 0; n < NTILES; ++n) {
        int N = n * 16 + ml;
        b2v[n] = b2[N];
        bav[n] = ba[N];
        wmv[n] = Wm[N];
        wbv[n] = Wb[N];
    }

    const int NT = N_EDGES / 16;
    for (int t = blockIdx.x * 4 + wv; t < NT; t += gridDim.x * 4) {
        int e0 = t * 16;

        // ---- GEMM2: relu(BN(hpre)) @ W2 + b2 ----
        v8f acc[NTILES];
        #pragma unroll
        for (int n = 0; n < NTILES; ++n) {
            #pragma unroll
            for (int r = 0; r < 8; ++r) acc[n][r] = b2v[n];
        }
        const float* hrow = hpre + (size_t)(e0 + ml) * HID;
        {
            Frag16 bb[2];
            ldB(bb[0], w2l, 0, 0, lane);
            #pragma unroll
            for (int c = 0; c < KCHUNKS; ++c) {
                Frag16 a;
                #pragma unroll
                for (int g = 0; g < 2; ++g) {
                    int c0 = c * 32 + g * 16 + half * 8;
                    float4 v0 = *(const float4*)(hrow + c0);
                    float4 v1 = *(const float4*)(hrow + c0 + 4);
                    float f[8] = {v0.x, v0.y, v0.z, v0.w, v1.x, v1.y, v1.z, v1.w};
                    #pragma unroll
                    for (int q = 0; q < 8; ++q) {
                        float sv = sc_l[c0 + q] * f[q] + bi_l[c0 + q];
                        a.s[g * 8 + q] = f2bf(fmaxf(sv, 0.f));
                    }
                }
                #pragma unroll
                for (int n = 0; n < NTILES; ++n) {
                    int idx = c * NTILES + n;
                    if (idx + 1 < KCHUNKS * NTILES)
                        ldB(bb[(idx + 1) & 1], w2l, (idx + 1) >> 3, (idx + 1) & 7, lane);
                    acc[n] = __builtin_amdgcn_wmma_f32_16x16x32_bf16(
                        false, a.v, false, bb[idx & 1].v, (short)0, acc[n], false, false);
                }
            }
        }

        // ---- ReLU + sigmoid gate: m_ij *= sigmoid(m_ij . Wm + bm) ----
        #pragma unroll
        for (int r = 0; r < 8; ++r) {
            float p = 0.f;
            #pragma unroll
            for (int n = 0; n < NTILES; ++n) {
                float m = fmaxf(acc[n][r], 0.f);
                acc[n][r] = m;
                p += m * wmv[n];
            }
            p += __shfl_xor(p, 1, 32);
            p += __shfl_xor(p, 2, 32);
            p += __shfl_xor(p, 4, 32);
            p += __shfl_xor(p, 8, 32);
            float w = sigmoid_f(p + bmv);
            #pragma unroll
            for (int n = 0; n < NTILES; ++n) acc[n][r] *= w;
        }

        // ---- C-layout -> A-layout transpose via per-wave LDS tile (bf16) ----
        #pragma unroll
        for (int n = 0; n < NTILES; ++n) {
            int N = n * 16 + ml;
            #pragma unroll
            for (int r = 0; r < 8; ++r)
                mt[wv][(half * 8 + r) * HID + N] = f2bf(acc[n][r]);
        }
        asm volatile("s_wait_dscnt 0" ::: "memory");

        // ---- GEMM3: relu(m_ij @ Wa + ba), then s = . @ Wb ----
        v8f acc3[NTILES];
        #pragma unroll
        for (int n = 0; n < NTILES; ++n) {
            #pragma unroll
            for (int r = 0; r < 8; ++r) acc3[n][r] = bav[n];
        }
        {
            Frag16 bb[2];
            ldB(bb[0], wal, 0, 0, lane);
            #pragma unroll
            for (int c = 0; c < KCHUNKS; ++c) {
                Frag16 a;
                a.u[0] = *(const uint4*)&mt[wv][ml * HID + c * 32 + half * 8];
                a.u[1] = *(const uint4*)&mt[wv][ml * HID + c * 32 + 16 + half * 8];
                #pragma unroll
                for (int n = 0; n < NTILES; ++n) {
                    int idx = c * NTILES + n;
                    if (idx + 1 < KCHUNKS * NTILES)
                        ldB(bb[(idx + 1) & 1], wal, (idx + 1) >> 3, (idx + 1) & 7, lane);
                    acc3[n] = __builtin_amdgcn_wmma_f32_16x16x32_bf16(
                        false, a.v, false, bb[idx & 1].v, (short)0, acc3[n], false, false);
                }
            }
        }
        #pragma unroll
        for (int r = 0; r < 8; ++r) {
            float p = 0.f;
            #pragma unroll
            for (int n = 0; n < NTILES; ++n) p += fmaxf(acc3[n][r], 0.f) * wbv[n];
            p += __shfl_xor(p, 1, 32);
            p += __shfl_xor(p, 2, 32);
            p += __shfl_xor(p, 4, 32);
            p += __shfl_xor(p, 8, 32);
            if (ml == 0) sS[wv][half * 8 + r] = p;
        }
        asm volatile("s_wait_dscnt 0" ::: "memory");

        // ---- scatter: x_next[row] += clip(x_diff * s, +/-100) ----
        for (int m = 0; m < 16; ++m) {
            int ee = e0 + m;
            int ri = row[ee], ci = col[ee];
            float sv = sS[wv][m];
            const float* xi = xc + (size_t)ri * HID;
            const float* xj = xc + (size_t)ci * HID;
            float* dst = xnxt + (size_t)ri * HID;
            #pragma unroll
            for (int c = lane; c < HID; c += 32) {
                float d  = xi[c] - xj[c];
                float tr = fminf(fmaxf(d * sv, -100.f), 100.f);
                atomicAdd(dst + c, tr);
            }
        }
    }
}

// ---------------------------------------------------------------------------
// Output head: sigmoid(x @ We + be), [N,128] @ [128,2].
// ---------------------------------------------------------------------------
__global__ void k_out(const float* __restrict__ x, const float* __restrict__ We,
                      const float* __restrict__ be, float* __restrict__ out) {
    int n = blockIdx.x * blockDim.x + threadIdx.x;
    if (n >= N_NODES) return;
    const float* xr = x + (size_t)n * HID;
    float d0 = 0.f, d1 = 0.f;
    #pragma unroll 4
    for (int k = 0; k < HID; ++k) {
        float v = xr[k];
        d0 += v * We[k * 2 + 0];
        d1 += v * We[k * 2 + 1];
    }
    out[n * 2 + 0] = sigmoid_f(d0 + be[0]);
    out[n * 2 + 1] = sigmoid_f(d1 + be[1]);
}

// ---------------------------------------------------------------------------
extern "C" void kernel_launch(void* const* d_in, const int* in_sizes, int n_in,
                              void* d_out, int out_size, void* d_ws, size_t ws_size,
                              hipStream_t stream) {
    (void)in_sizes; (void)n_in; (void)out_size; (void)ws_size;
    const float* x_in  = (const float*)d_in[0];
    const int*   eidx  = (const int*)  d_in[1];
    const float* eattr = (const float*)d_in[2];
    const float* W1    = (const float*)d_in[3];
    const float* gamma = (const float*)d_in[4];
    const float* beta  = (const float*)d_in[5];
    const float* W2    = (const float*)d_in[6];
    const float* b2    = (const float*)d_in[7];
    const float* Wa    = (const float*)d_in[8];
    const float* ba    = (const float*)d_in[9];
    const float* Wb    = (const float*)d_in[10];
    const float* Wm    = (const float*)d_in[11];
    const float* bm    = (const float*)d_in[12];
    const float* We    = (const float*)d_in[13];
    const float* be    = (const float*)d_in[14];
    const int* row = eidx;
    const int* col = eidx + N_EDGES;

    char* ws = (char*)d_ws;
    size_t off = 0;
    auto alloc = [&](size_t bytes) -> void* {
        void* p = ws + off;
        off += (bytes + 255) & ~(size_t)255;
        return p;
    };
    float*          xb0   = (float*)alloc((size_t)N_NODES * HID * 4);
    float*          xb1   = (float*)alloc((size_t)N_NODES * HID * 4);
    unsigned short* xbf   = (unsigned short*)alloc((size_t)N_NODES * HID * 2);
    float*          stats = (float*)alloc(512 * 4);
    unsigned short* wsw   = (unsigned short*)alloc((size_t)3 * 4 * WMAT * 2);
    float*          hpre  = (float*)alloc((size_t)N_EDGES * HID * 4);

    float* stat_sum = stats;
    float* stat_sq  = stats + 128;
    float* scale    = stats + 256;
    float* bias     = stats + 384;

    k_prep_weights<<<(3 * 4 * WMAT + 255) / 256, 256, 0, stream>>>(W1, W2, Wa, wsw);

    const float* cur = x_in;
    for (int l = 0; l < 3; ++l) {
        float* nxt = (l & 1) ? xb1 : xb0;
        k_copy<<<10000, 256, 0, stream>>>(cur, nxt, xbf, stats);
        const unsigned short* swl = wsw + (size_t)l * 4 * WMAT;
        k_edge_pass1<<<625, 128, 0, stream>>>(
            xbf, cur, row, col, eattr,
            swl, swl + WMAT,
            W1 + (size_t)l * D_IN * HID + 256 * HID,
            hpre, stat_sum, stat_sq);
        k_bn<<<1, 128, 0, stream>>>(stat_sum, stat_sq, gamma + l * HID, beta + l * HID,
                                    scale, bias);
        k_edge_pass2<<<1250, 128, 0, stream>>>(
            hpre, scale, bias,
            swl + 2 * WMAT, swl + 3 * WMAT,
            b2 + l * HID, ba + l * HID, Wm + l * HID, bm + l, Wb + l * HID,
            cur, row, col, nxt);
        cur = nxt;
    }
    k_out<<<(N_NODES + 255) / 256, 256, 0, stream>>>(cur, We, be, (float*)d_out);
}
